// SiglipEncoderLayer_57148834840669
// MI455X (gfx1250) — compile-verified, matrix operand order checked by reading, not run on hardware
//
#include <hip/hip_runtime.h>
#include <hip/hip_bf16.h>
#include <math.h>

// ---------------------------------------------------------------------------
// DiffAttention encoder layer for gfx1250 (MI455X), wave32 + WMMA bf16.
// All GEMMs run on v_wmma_f32_16x16x32_bf16 (f32 accumulate).
// Workspace requirement: ~285 MB.
// ---------------------------------------------------------------------------

typedef __bf16 bf16_t;
typedef __attribute__((ext_vector_type(16))) __bf16 v16bf;
typedef __attribute__((ext_vector_type(8)))  float  v8f;

#define LAMBDA_INIT_F           0.77787009955930822f
#define ONE_MINUS_LAMBDA_INIT_F 0.22212990044069178f

#define MB   8192   // B*S tokens
#define ED   1152   // hidden
#define FFD  4304   // intermediate
#define FFP  4352   // padded intermediate (multiple of 64)
#define SD   256    // seq len
#define BD   32     // batch
#define HN   16     // heads
#define HDIM 72     // head dim
#define HD2  36     // differential half head dim

__device__ __forceinline__ v8f wmma_bf16(v16bf a, v16bf b, v8f c) {
  // D = A(16x32) * B(32x16) + C, f32 accumulate
  return __builtin_amdgcn_wmma_f32_16x16x32_bf16(false, a, false, b, (short)0, c,
                                                 false, false);
}

__device__ __forceinline__ float gelu_tanh(float x) {
  float x3 = x * x * x;
  return 0.5f * x * (1.0f + tanhf(0.7978845608028654f * (x + 0.044715f * x3)));
}

// ---------------------------------------------------------------------------
// lambda_full = exp(sum lq1*lk1) - exp(sum lq2*lk2) + LAMBDA_INIT
// ---------------------------------------------------------------------------
__global__ void lambda_kernel(const float* __restrict__ lq1, const float* __restrict__ lk1,
                              const float* __restrict__ lq2, const float* __restrict__ lk2,
                              float* __restrict__ out) {
  int t = threadIdx.x;                 // 64 threads
  float p1 = (t < HD2) ? lq1[t] * lk1[t] : 0.f;
  float p2 = (t < HD2) ? lq2[t] * lk2[t] : 0.f;
  for (int off = 16; off; off >>= 1) {
    p1 += __shfl_xor(p1, off, 32);
    p2 += __shfl_xor(p2, off, 32);
  }
  __shared__ float sh[4];
  if ((t & 31) == 0) { sh[(t >> 5) * 2] = p1; sh[(t >> 5) * 2 + 1] = p2; }
  __syncthreads();
  if (t == 0)
    out[0] = __expf(sh[0] + sh[2]) - __expf(sh[1] + sh[3]) + LAMBDA_INIT_F;
}

// ---------------------------------------------------------------------------
// f32 -> bf16 cast with zero padding (rows_dst x cols_dst from rows_src x cols_src)
// ---------------------------------------------------------------------------
__global__ void pad_cast_kernel(const float* __restrict__ src, bf16_t* __restrict__ dst,
                                int rows_dst, int cols_dst, int rows_src, int cols_src) {
  size_t total = (size_t)rows_dst * cols_dst;
  for (size_t idx = (size_t)blockIdx.x * 256 + threadIdx.x; idx < total;
       idx += (size_t)gridDim.x * 256) {
    int r = (int)(idx / cols_dst);
    int c = (int)(idx - (size_t)r * cols_dst);
    float v = (r < rows_src && c < cols_src) ? src[(size_t)r * cols_src + c] : 0.f;
    dst[idx] = (bf16_t)v;
  }
}

__global__ void zero16_kernel(uint4* __restrict__ p, size_t n16) {
  uint4 z = {0u, 0u, 0u, 0u};
  for (size_t i = (size_t)blockIdx.x * 256 + threadIdx.x; i < n16;
       i += (size_t)gridDim.x * 256)
    p[i] = z;
}

// ---------------------------------------------------------------------------
// Row RMSNorm + cast to bf16:   out = x * rsqrt(mean(x^2)+eps) * w
// ---------------------------------------------------------------------------
__global__ __launch_bounds__(256)
void rmsnorm_cast_kernel(const float* __restrict__ x, const float* __restrict__ w,
                         bf16_t* __restrict__ out, int ncols, float eps) {
  const int row = blockIdx.x;
  const float* xr = x + (size_t)row * ncols;
  float ss = 0.f;
  for (int i = threadIdx.x; i < ncols; i += 256) { float v = xr[i]; ss += v * v; }
  for (int off = 16; off; off >>= 1) ss += __shfl_xor(ss, off, 32);
  __shared__ float red[8];
  if ((threadIdx.x & 31) == 0) red[threadIdx.x >> 5] = ss;
  __syncthreads();
  if (threadIdx.x == 0) {
    float tot = 0.f;
    for (int i = 0; i < 8; ++i) tot += red[i];
    red[0] = tot;
  }
  __syncthreads();
  const float rinv = rsqrtf(red[0] / (float)ncols + eps);
  for (int i = threadIdx.x; i < ncols; i += 256)
    out[(size_t)row * ncols + i] = (bf16_t)(xr[i] * rinv * w[i]);
}

// ---------------------------------------------------------------------------
// Generic WMMA GEMM:  C[M][N] = A[M][KT](bf16) * Bw[N][KT]^T(bf16) + bias (+res)
// KT is compile-time so all tile strides fold into the 24-bit VMEM immediate
// offsets -> inner loop is 2 pointer increments + 12 b128 loads + 8 WMMAs.
// Block = 256 threads (8 waves). Each wave: 32x64 output (2x4 WMMA tiles).
// grid.x = N/64 chunks, grid.y = M/256 chunks (8 waves * 32 rows).
// SMODE: 0 = linear [M][ldc], 1 = scatter to q/k pad [B][2H][S][64],
//        2 = scatter to vT pad [B][H][80][S]
// ---------------------------------------------------------------------------
template <int SMODE, int ACT, int RES, int OBF, int KT>
__global__ __launch_bounds__(256)
void gemm_wmma_kernel(const bf16_t* __restrict__ A, const bf16_t* __restrict__ Bw,
                      const float* __restrict__ bias, const float* __restrict__ res,
                      void* __restrict__ outp, int Nlog, int ldc) {
  const int lane = threadIdx.x & 31;
  const int wv   = threadIdx.x >> 5;
  const int m0   = blockIdx.y * 256 + wv * 32;
  const int n0   = blockIdx.x * 64;
  const int lm   = lane & 15;          // row-in-tile (A) / col-in-tile (B,C)
  const int kh   = (lane >> 4) * 16;   // K half selector
  const int rsel = (lane >> 4) * 8;    // C row-group selector

  v8f acc[2][4] = {};
  const bf16_t* ap = A  + (size_t)(m0 + lm) * KT + kh;
  const bf16_t* bp = Bw + (size_t)(n0 + lm) * KT + kh;

#pragma unroll 2
  for (int kk = 0; kk < KT; kk += 32) {
    if (kk + 256 < KT) {  // gfx1250 global_prefetch, constant distance
      __builtin_prefetch((const void*)(ap + 256), 0, 1);
      __builtin_prefetch((const void*)(bp + 256), 0, 1);
    }
    v16bf a0 = *(const v16bf*)(ap);
    v16bf a1 = *(const v16bf*)(ap + (size_t)16 * KT);
    v16bf b0 = *(const v16bf*)(bp);
    v16bf b1 = *(const v16bf*)(bp + (size_t)16 * KT);
    v16bf b2 = *(const v16bf*)(bp + (size_t)32 * KT);
    v16bf b3 = *(const v16bf*)(bp + (size_t)48 * KT);
    acc[0][0] = wmma_bf16(a0, b0, acc[0][0]);
    acc[0][1] = wmma_bf16(a0, b1, acc[0][1]);
    acc[0][2] = wmma_bf16(a0, b2, acc[0][2]);
    acc[0][3] = wmma_bf16(a0, b3, acc[0][3]);
    acc[1][0] = wmma_bf16(a1, b0, acc[1][0]);
    acc[1][1] = wmma_bf16(a1, b1, acc[1][1]);
    acc[1][2] = wmma_bf16(a1, b2, acc[1][2]);
    acc[1][3] = wmma_bf16(a1, b3, acc[1][3]);
    ap += 32;
    bp += 32;
  }

#pragma unroll
  for (int mt = 0; mt < 2; ++mt)
#pragma unroll
    for (int nt = 0; nt < 4; ++nt) {
      const int n = n0 + nt * 16 + lm;
      const float bvv = (n < Nlog) ? bias[n] : 0.f;
#pragma unroll
      for (int r = 0; r < 8; ++r) {
        const int m = m0 + mt * 16 + r + rsel;
        float x = acc[mt][nt][r] + bvv;
        if (SMODE == 0) {
          if (ACT) x = (n < Nlog) ? gelu_tanh(x) : 0.f;
          if (RES) x += res[(size_t)m * ldc + n];
          if (OBF) ((bf16_t*)outp)[(size_t)m * ldc + n] = (bf16_t)x;
          else     ((float*)outp)[(size_t)m * ldc + n] = x;
        } else if (SMODE == 1) {      // q/k: [B][2H][S][64] (K padded 36->64)
          const int dh = n / HD2, d = n - dh * HD2;
          const int b = m / SD, s = m - b * SD;
          ((bf16_t*)outp)[(((size_t)(b * 2 * HN + dh)) * SD + s) * 64 + d] = (bf16_t)x;
        } else {                      // v^T: [B][H][80][S] (N padded 72->80)
          const int hh = n / HDIM, d = n - hh * HDIM;
          const int b = m / SD, s = m - b * SD;
          ((bf16_t*)outp)[(((size_t)(b * HN + hh)) * 80 + d) * SD + s] = (bf16_t)x;
        }
      }
    }
}

// ---------------------------------------------------------------------------
// Differential attention for one (batch, head, 32-query-row) block.
// 8 waves. Scores via WMMA into 64KB LDS, per-row softmax, combine s1-λs2,
// A·V^T via WMMA (A operand from LDS), sub-LN, scaled bf16 writeout.
// ---------------------------------------------------------------------------
__global__ __launch_bounds__(256)
void diffattn_kernel(const bf16_t* __restrict__ qpad, const bf16_t* __restrict__ kpad,
                     const bf16_t* __restrict__ vtp, const float* __restrict__ sublnw,
                     const float* __restrict__ lambdaPtr, bf16_t* __restrict__ attn_out) {
  __shared__ float s1[32][256];   // comp-0 scores / later bf16 A matrix
  __shared__ float s2[32][256];   // comp-1 scores / later f32 AV output

  const int sq0  = blockIdx.x * 32;
  const int h    = blockIdx.y;
  const int b    = blockIdx.z;
  const int lane = threadIdx.x & 31;
  const int wv   = threadIdx.x >> 5;
  const int lm   = lane & 15;
  const int kh   = (lane >> 4) * 16;
  const int rsel = (lane >> 4) * 8;
  const float scale = rsqrtf((float)HDIM);   // reference scales by sqrt(HD)

  // ---- Phase A: raw scores for both dual components (64 16x16 tiles) ----
#pragma unroll
  for (int t = 0; t < 8; ++t) {
    const int tile = wv * 8 + t;
    const int comp = tile >> 5;
    const int rem  = tile & 31;
    const int qt   = rem >> 4;
    const int kt   = rem & 15;
    const int dh   = 2 * h + comp;
    const bf16_t* qb =
        qpad + (((size_t)(b * 2 * HN + dh)) * SD + (sq0 + qt * 16 + lm)) * 64 + kh;
    const bf16_t* kb =
        kpad + (((size_t)(b * 2 * HN + dh)) * SD + (kt * 16 + lm)) * 64 + kh;
    v8f acc = {};
    acc = wmma_bf16(*(const v16bf*)qb, *(const v16bf*)kb, acc);
    acc = wmma_bf16(*(const v16bf*)(qb + 32), *(const v16bf*)(kb + 32), acc);
    float* dst = comp ? &s2[0][0] : &s1[0][0];
#pragma unroll
    for (int r = 0; r < 8; ++r)
      dst[(qt * 16 + r + rsel) * 256 + kt * 16 + lm] = acc[r] * scale;
  }
  __syncthreads();

  // ---- Phase B: row softmax (each warp owns 4 rows) ----
#pragma unroll
  for (int rr = 0; rr < 4; ++rr) {
    const int row = wv * 4 + rr;
#pragma unroll
    for (int c = 0; c < 2; ++c) {
      float* srow = c ? s2[row] : s1[row];
      float vals[8];
      float mx = -INFINITY;
#pragma unroll
      for (int i = 0; i < 8; ++i) { vals[i] = srow[lane + i * 32]; mx = fmaxf(mx, vals[i]); }
      for (int off = 16; off; off >>= 1) mx = fmaxf(mx, __shfl_xor(mx, off, 32));
      float sum = 0.f;
#pragma unroll
      for (int i = 0; i < 8; ++i) { vals[i] = __expf(vals[i] - mx); sum += vals[i]; }
      for (int off = 16; off; off >>= 1) sum += __shfl_xor(sum, off, 32);
      const float inv = 1.0f / sum;
#pragma unroll
      for (int i = 0; i < 8; ++i) srow[lane + i * 32] = vals[i] * inv;
    }
  }
  __syncthreads();

  // ---- Phase C: A = s1 - lambda*s2, packed bf16 over s1's storage ----
  const float lambda = *lambdaPtr;
  float av[32];
  {
    const int row = threadIdx.x >> 3;
    const int c0  = (threadIdx.x & 7) * 32;
#pragma unroll
    for (int i = 0; i < 32; ++i) av[i] = s1[row][c0 + i] - lambda * s2[row][c0 + i];
  }
  __syncthreads();
  bf16_t* aM = (bf16_t*)&s1[0][0];  // [32][256] bf16, aliases s1
  {
    const int row = threadIdx.x >> 3;
    const int c0  = (threadIdx.x & 7) * 32;
#pragma unroll
    for (int i = 0; i < 32; ++i) aM[row * 256 + c0 + i] = (bf16_t)av[i];
  }
  __syncthreads();

  // ---- Phase D: out[32][80] = A[32][256] x V^T (10 tiles over 8 waves) ----
  float* oM = &s2[0][0];  // [32][80] f32, aliases s2
  for (int t = wv; t < 10; t += 8) {
    const int mt = t / 5, nt = t % 5;
    v8f acc = {};
    const bf16_t* vb =
        vtp + (((size_t)(b * HN + h)) * 80 + (nt * 16 + lm)) * SD + kh;
    const bf16_t* ab = aM + (mt * 16 + lm) * 256 + kh;
#pragma unroll
    for (int kk = 0; kk < 256; kk += 32)
      acc = wmma_bf16(*(const v16bf*)(ab + kk), *(const v16bf*)(vb + kk), acc);
#pragma unroll
    for (int r = 0; r < 8; ++r)
      oM[(mt * 16 + r + rsel) * 80 + nt * 16 + lm] = acc[r];
  }
  __syncthreads();

  // ---- Phase E: sub-LN RMSNorm over head dim + (1-lambda_init) scale ----
  {
    const int row = threadIdx.x >> 3;   // 0..31
    const int g   = threadIdx.x & 7;    // 8 threads per row, 9 elems each
    float vals[9];
    float ss = 0.f;
#pragma unroll
    for (int i = 0; i < 9; ++i) {
      const float x = oM[row * 80 + g * 9 + i];
      vals[i] = x;
      ss += x * x;
    }
    ss += __shfl_xor(ss, 1, 32);
    ss += __shfl_xor(ss, 2, 32);
    ss += __shfl_xor(ss, 4, 32);
    const float mul = rsqrtf(ss / (float)HDIM + 1e-5f) * ONE_MINUS_LAMBDA_INIT_F;
    const int s = sq0 + row;
    const size_t obase = ((size_t)(b * SD + s)) * ED + h * HDIM;
#pragma unroll
    for (int i = 0; i < 9; ++i) {
      const int d = g * 9 + i;
      attn_out[obase + d] = (bf16_t)(vals[i] * mul * sublnw[d]);
    }
  }
}

// ---------------------------------------------------------------------------
extern "C" void kernel_launch(void* const* d_in, const int* in_sizes, int n_in,
                              void* d_out, int out_size, void* d_ws, size_t ws_size,
                              hipStream_t stream) {
  (void)in_sizes; (void)n_in; (void)out_size; (void)ws_size;
  const float* hs     = (const float*)d_in[0];
  const float* Wq     = (const float*)d_in[1];
  const float* bq     = (const float*)d_in[2];
  const float* Wk     = (const float*)d_in[3];
  const float* bk     = (const float*)d_in[4];
  const float* Wv     = (const float*)d_in[5];
  const float* bv     = (const float*)d_in[6];
  const float* Wo     = (const float*)d_in[7];
  const float* bo     = (const float*)d_in[8];
  const float* lq1    = (const float*)d_in[9];
  const float* lk1    = (const float*)d_in[10];
  const float* lq2    = (const float*)d_in[11];
  const float* lk2    = (const float*)d_in[12];
  const float* sublnw = (const float*)d_in[13];
  const float* rms1   = (const float*)d_in[14];
  const float* rms2   = (const float*)d_in[15];
  const float* fc1w   = (const float*)d_in[16];
  const float* fc1b   = (const float*)d_in[17];
  const float* fc2w   = (const float*)d_in[18];
  const float* fc2b   = (const float*)d_in[19];

  // ----- workspace layout (all offsets 256B-aligned) -----
  char* ws = (char*)d_ws;
  size_t off = 0;
  auto alloc = [&](size_t bytes) -> void* {
    void* p = ws + off;
    off += (bytes + 255) & ~(size_t)255;
    return p;
  };
  float*  lam   = (float*) alloc(4);
  bf16_t* xn    = (bf16_t*)alloc((size_t)MB * ED * 2);     // rms1(x) bf16
  bf16_t* wqb   = (bf16_t*)alloc((size_t)ED * ED * 2);
  bf16_t* wkb   = (bf16_t*)alloc((size_t)ED * ED * 2);
  bf16_t* wvb   = (bf16_t*)alloc((size_t)ED * ED * 2);
  bf16_t* wob   = (bf16_t*)alloc((size_t)ED * ED * 2);
  bf16_t* fc1bf = (bf16_t*)alloc((size_t)FFP * ED * 2);    // rows padded w/ zeros
  bf16_t* fc2bf = (bf16_t*)alloc((size_t)ED * FFP * 2);    // K cols padded w/ zeros
  bf16_t* qpad  = (bf16_t*)alloc((size_t)BD * 2 * HN * SD * 64 * 2);
  bf16_t* kpad  = (bf16_t*)alloc((size_t)BD * 2 * HN * SD * 64 * 2);
  bf16_t* vtp   = (bf16_t*)alloc((size_t)BD * HN * 80 * SD * 2);
  bf16_t* attnb = (bf16_t*)alloc((size_t)MB * ED * 2);     // attention out bf16
  float*  hbuf  = (float*) alloc((size_t)MB * ED * 4);     // residual1 + attn
  bf16_t* ybuf  = (bf16_t*)alloc((size_t)MB * ED * 2);     // rms2(h) bf16
  bf16_t* actb  = (bf16_t*)alloc((size_t)MB * FFP * 2);    // gelu(fc1) bf16

  const dim3 blk(256);

  // scalar lambda
  lambda_kernel<<<1, 64, 0, stream>>>(lq1, lk1, lq2, lk2, lam);

  // weight conversion (bf16, zero padded)
  pad_cast_kernel<<<2048, blk, 0, stream>>>(Wq,   wqb,   ED,  ED,  ED,  ED);
  pad_cast_kernel<<<2048, blk, 0, stream>>>(Wk,   wkb,   ED,  ED,  ED,  ED);
  pad_cast_kernel<<<2048, blk, 0, stream>>>(Wv,   wvb,   ED,  ED,  ED,  ED);
  pad_cast_kernel<<<2048, blk, 0, stream>>>(Wo,   wob,   ED,  ED,  ED,  ED);
  pad_cast_kernel<<<4096, blk, 0, stream>>>(fc1w, fc1bf, FFP, ED,  FFD, ED);
  pad_cast_kernel<<<4096, blk, 0, stream>>>(fc2w, fc2bf, ED,  FFP, ED,  FFD);

  // zero the padded q/k/vT region (contiguous: qpad..vtp)
  {
    size_t bytes = (size_t)(BD * 2 * HN * SD * 64 * 2) * 2 + (size_t)BD * HN * 80 * SD * 2;
    zero16_kernel<<<4096, blk, 0, stream>>>((uint4*)qpad, bytes / 16);
  }

  // x_norm = RMSNorm(hidden, rms1) -> bf16
  rmsnorm_cast_kernel<<<MB, blk, 0, stream>>>(hs, rms1, xn, ED, 1e-6f);

  // Q/K/V projections (scatter-stored into padded attention layouts)
  gemm_wmma_kernel<1, 0, 0, 1, ED><<<dim3(ED / 64, MB / 256), blk, 0, stream>>>(
      xn, wqb, bq, nullptr, qpad, ED, ED);
  gemm_wmma_kernel<1, 0, 0, 1, ED><<<dim3(ED / 64, MB / 256), blk, 0, stream>>>(
      xn, wkb, bk, nullptr, kpad, ED, ED);
  gemm_wmma_kernel<2, 0, 0, 1, ED><<<dim3(ED / 64, MB / 256), blk, 0, stream>>>(
      xn, wvb, bv, nullptr, vtp, ED, ED);

  // differential attention
  diffattn_kernel<<<dim3(SD / 32, HN, BD), blk, 0, stream>>>(
      qpad, kpad, vtp, sublnw, lam, attnb);

  // h = hidden + attn @ Wo^T + bo   (f32)
  gemm_wmma_kernel<0, 0, 1, 0, ED><<<dim3(ED / 64, MB / 256), blk, 0, stream>>>(
      attnb, wob, bo, hs, hbuf, ED, ED);

  // y = RMSNorm(h, rms2) -> bf16
  rmsnorm_cast_kernel<<<MB, blk, 0, stream>>>(hbuf, rms2, ybuf, ED, 1e-6f);

  // act = gelu(y @ fc1^T + b1) -> bf16, padded cols zeroed
  gemm_wmma_kernel<0, 1, 0, 1, ED><<<dim3(FFP / 64, MB / 256), blk, 0, stream>>>(
      ybuf, fc1bf, fc1b, nullptr, actb, FFD, FFP);

  // out = h + act @ fc2^T + b2   (f32)
  gemm_wmma_kernel<0, 0, 1, 0, FFP><<<dim3(ED / 64, MB / 256), blk, 0, stream>>>(
      actb, fc2bf, fc2b, hbuf, d_out, ED, ED);
}
